// BaseModel_87179246174217
// MI455X (gfx1250) — matrix operand order repaired and checked by reading, not compiled, not run on hardware
//
#include <hip/hip_runtime.h>
#include <math.h>

// ---------------- constants from the reference ----------------
#define C_C        3.0f
#define K_K        (1.0f / 3.0f)
#define SQRT_K     0.57735026918962576f
#define GAMMA      0.1f
#define LAMBDA_S   5.0f
#define EPS_       1e-7f
#define MIN_NORM   1e-15f
#define MAX_SQD    50.0f
#define N_NODES    100000
#define HID        64
#define XD         65          // hyperboloid dim = HID+1
#define E_POS      250000
#define E_NEG      250000
#define E_NONE     500000
#define E_TOT      1000000     // pos + neg + none2 for info net

// info net matmul tiling
#define KPAD       160         // 131 -> 160 (5 x K32)
#define NPAD       144         // 130 -> 144 (9 x N16)
#define NT         9
#define KT         5
#define MT         2           // 16-row M-subtiles per wave (share B fragments)
#define WAVES      2           // waves per block
#define WTILES     (E_TOT / (16 * MT))     // 31250 wave tiles (32 edges each)
#define NSUB       (E_TOT / 16)            // 62500 partial slots
#define INFO_BLOCKS (WTILES / WAVES)       // 15625

typedef __bf16 bf16_t;
typedef __attribute__((ext_vector_type(16))) __bf16       v16bf;
typedef __attribute__((ext_vector_type(8)))  float        v8f;
typedef __attribute__((ext_vector_type(8)))  unsigned int v8u;

__device__ __forceinline__ unsigned short f2bf(float f) {
    unsigned int u = __builtin_bit_cast(unsigned int, f);
    unsigned int r = (u + 0x7FFFu + ((u >> 16) & 1u)) >> 16;   // RNE
    return (unsigned short)r;
}

// ---------------- 1) expmap0 + proj: z -> x (f32 and bf16 copies) -------------
__global__ __launch_bounds__(256) void expmap_kernel(const float* __restrict__ z,
                                                     float* __restrict__ xf,
                                                     unsigned short* __restrict__ xbf) {
    int wave = blockIdx.x * 8 + (threadIdx.x >> 5);
    int lane = threadIdx.x & 31;
    if (wave >= N_NODES) return;
    const float* zr = z + (size_t)wave * HID;
    float a = zr[lane], b = zr[lane + 32];
    float ss = a * a + b * b;
    #pragma unroll
    for (int off = 16; off > 0; off >>= 1) ss += __shfl_down(ss, off, 32);
    ss = __shfl(ss, 0, 32);
    float xn    = fmaxf(sqrtf(ss), MIN_NORM);
    float theta = xn / SQRT_K;
    float sc    = SQRT_K * sinhf(theta) / xn;      // y = sc * z
    float y1 = sc * a, y2 = sc * b;
    float x0 = sqrtf(K_K + sc * sc * ss);          // proj: sqrt(K + ||y||^2)
    float* xr = xf + (size_t)wave * XD;
    unsigned short* xb = xbf + (size_t)wave * XD;
    if (lane == 0) { xr[0] = x0; xb[0] = f2bf(x0); }
    xr[1 + lane]  = y1;  xr[33 + lane] = y2;
    xb[1 + lane]  = f2bf(y1);  xb[33 + lane] = f2bf(y2);
}

// ---------------- 2) w1 -> bf16 padded [NPAD][KPAD] ----------------
__global__ __launch_bounds__(256) void w1prep_kernel(const float* __restrict__ w1,
                                                     unsigned short* __restrict__ w1p) {
    int t = blockIdx.x * 256 + threadIdx.x;
    if (t >= NPAD * KPAD) return;
    int n = t / KPAD, k = t % KPAD;
    unsigned short v = 0;
    if (n < 130 && k < 131) v = f2bf(w1[n * 131 + k]);
    w1p[t] = v;
}

// ---------------- 3) nll for one edge set / target class ----------------
__global__ __launch_bounds__(256) void nll_kernel(const float* __restrict__ z,
                                                  const float* __restrict__ lin_w,
                                                  const float* __restrict__ lin_b,
                                                  const int* __restrict__ ei, int E, int cls,
                                                  float* __restrict__ partials) {
    __shared__ float red[256];
    int t = blockIdx.x * 256 + threadIdx.x;
    float val = 0.f;
    if (t < E) {
        int i = ei[t], j = ei[E + t];
        const float* zi = z + (size_t)i * HID;
        const float* zj = z + (size_t)j * HID;
        float l0 = lin_b[0], l1 = lin_b[1], l2 = lin_b[2];
        #pragma unroll 8
        for (int k = 0; k < HID; ++k) {
            float a = zi[k], b = zj[k];
            l0 += lin_w[k] * a + lin_w[64 + k]  * b;
            l1 += lin_w[128 + k] * a + lin_w[192 + k] * b;
            l2 += lin_w[256 + k] * a + lin_w[320 + k] * b;
        }
        float m   = fmaxf(l0, fmaxf(l1, l2));
        float lse = m + logf(expf(l0 - m) + expf(l1 - m) + expf(l2 - m));
        float lc  = (cls == 0) ? l0 : ((cls == 1) ? l1 : l2);
        val = lse - lc;
    }
    red[threadIdx.x] = val;
    __syncthreads();
    #pragma unroll
    for (int s = 128; s > 0; s >>= 1) {
        if (threadIdx.x < s) red[threadIdx.x] += red[threadIdx.x + s];
        __syncthreads();
    }
    if (threadIdx.x == 0) partials[blockIdx.x] = red[0];
}

// ---------------- 4) triplet losses (l1: sign=+1 on pos; l2: sign=-1 on neg) --
__device__ __forceinline__ float hyp_sqdist_dev(const float* __restrict__ a,
                                                const float* __restrict__ b) {
    float dot = 0.f;
    #pragma unroll 13
    for (int d = 0; d < XD; ++d) dot += a[d] * b[d];
    float mdot = dot - 2.f * a[0] * b[0];
    float th   = fmaxf(-mdot * C_C, 1.f + EPS_);
    float ac   = logf(th + sqrtf(th * th - 1.f));
    return fminf(K_K * ac * ac, MAX_SQD);
}

__global__ __launch_bounds__(256) void triplet_kernel(const float* __restrict__ xf,
                                                      const int* __restrict__ ei,
                                                      const int* __restrict__ kk,
                                                      int E, float sign,
                                                      float* __restrict__ partials) {
    __shared__ float red[256];
    int t = blockIdx.x * 256 + threadIdx.x;
    float val = 0.f;
    if (t < E) {
        int i = ei[t], j = ei[E + t], k = kk[t];
        float dij = hyp_sqdist_dev(xf + (size_t)i * XD, xf + (size_t)j * XD);
        float dik = hyp_sqdist_dev(xf + (size_t)i * XD, xf + (size_t)k * XD);
        val = fmaxf(sign * (dij - dik), 0.f);
    }
    red[threadIdx.x] = val;
    __syncthreads();
    #pragma unroll
    for (int s = 128; s > 0; s >>= 1) {
        if (threadIdx.x < s) red[threadIdx.x] += red[threadIdx.x + s];
        __syncthreads();
    }
    if (threadIdx.x == 0) partials[blockIdx.x] = red[0];
}

// ---------------- 5) info net: WMMA bf16 matmul + fused epilogue ----------------
// Each wave: 32 edges as 2 M-subtiles of 16 that SHARE every B fragment.
// A row (per edge): [x_i(65) | x_j(65) | y | 0-pad] padded to KPAD, bf16.
// B: w1 padded to [NPAD][KPAD] bf16 => base[m][n] = sum_k A[m][k]*w1[n][k]
// already includes y*w1[n][130] for the *pred* labels; the shuffle pass adds
// (y_perm - y)*w1[n][130] before the ReLU.
__global__ __launch_bounds__(32 * WAVES) void info_kernel(
        const unsigned short* __restrict__ xbf,
        const unsigned short* __restrict__ w1p,
        const float* __restrict__ info_w1,
        const float* __restrict__ b1,
        const float* __restrict__ w2,
        const float* __restrict__ b2,
        const int* __restrict__ pos,
        const int* __restrict__ neg,
        const int* __restrict__ none2,
        const int* __restrict__ perm,
        float* __restrict__ partial_pred,
        float* __restrict__ partial_exp) {
    __shared__ unsigned short As[WAVES][MT * 16][KPAD];  // 2*32*160*2 = 20 KB
    __shared__ float          Os[WAVES][16][NPAD];       // 2*16*144*4 = 18 KB
    int lane  = threadIdx.x & 31;
    int wave  = threadIdx.x >> 5;
    int wtile = blockIdx.x * WAVES + wave;
    if (wtile >= WTILES) return;                         // wave-uniform
    int ebase = wtile * (16 * MT);
    int r15   = lane & 15;
    int hi    = lane >> 4;

    // ---- stage 2x16 edge rows into LDS (lane pair covers one row) ----
    #pragma unroll
    for (int mt = 0; mt < MT; ++mt) {
        int e = ebase + mt * 16 + r15;
        int i, j; float y;
        if (e < E_POS)          { i = pos[e];  j = pos[E_POS + e];  y = 0.f; }
        else if (e < 2 * E_POS) { int q = e - E_POS;     i = neg[q];   j = neg[E_NEG + q];    y = 1.f; }
        else                    { int q = e - 2 * E_POS; i = none2[q]; j = none2[E_NONE + q]; y = 2.f; }
        const unsigned short* xi = xbf + (size_t)i * XD;
        const unsigned short* xj = xbf + (size_t)j * XD;
        int k0 = hi * (KPAD / 2);
        for (int kk2 = 0; kk2 < KPAD / 2; ++kk2) {
            int k = k0 + kk2;
            unsigned short v;
            if      (k < 65)   v = xi[k];
            else if (k < 130)  v = xj[k - 65];
            else if (k == 130) v = f2bf(y);
            else               v = 0;
            As[wave][mt * 16 + r15][k] = v;
        }
    }
    // same wave writes then reads its own LDS slice: DS ops are in-order per wave.

    // ---- (2x16) x 160-K x 144-N matmul: 90 WMMAs, B fragments shared ----
    v8f acc[MT][NT];
    #pragma unroll
    for (int mt = 0; mt < MT; ++mt)
        #pragma unroll
        for (int t = 0; t < NT; ++t)
            acc[mt][t] = (v8f){0.f,0.f,0.f,0.f,0.f,0.f,0.f,0.f};

    #pragma unroll
    for (int ks = 0; ks < KT; ++ks) {
        // A fragments: this code's contiguous k order, matching B (lanes 0-15: K 0..15; hi: +16)
        v16bf afrag[MT];
        #pragma unroll
        for (int mt = 0; mt < MT; ++mt) {
            v8u a_u;
            #pragma unroll
            for (int v = 0; v < 8; ++v) {
                int kh = ks * 32 + 2 * v + hi * 16;
                a_u[v] = *(const unsigned int*)&As[wave][mt * 16 + r15][kh];
            }
            afrag[mt] = __builtin_bit_cast(v16bf, a_u);
        }
        #pragma unroll
        for (int t = 0; t < NT; ++t) {
            // B fragment: lane n holds col N=t*16+n, K kb..kb+15 contiguous
            int n  = t * 16 + r15;
            int kb = ks * 32 + hi * 16;
            v8u b_u = *(const v8u*)&w1p[(size_t)n * KPAD + kb];
            v16bf bfrag = __builtin_bit_cast(v16bf, b_u);
            acc[0][t] = __builtin_amdgcn_wmma_f32_16x16x32_bf16(
                false, afrag[0], false, bfrag, (short)0, acc[0][t], false, false);
            acc[1][t] = __builtin_amdgcn_wmma_f32_16x16x32_bf16(
                false, afrag[1], false, bfrag, (short)0, acc[1][t], false, false);
        }
    }

    // ---- per-subtile: spill C to LDS, fused ReLU/w2/exp epilogue ----
    int rbase = hi * 8;
    #pragma unroll
    for (int mt = 0; mt < MT; ++mt) {
        #pragma unroll
        for (int t = 0; t < NT; ++t) {
            #pragma unroll
            for (int rr = 0; rr < 8; ++rr)
                Os[wave][rbase + rr][t * 16 + r15] = acc[mt][t][rr];
        }
        float sp = 0.f, se = 0.f;
        if (lane < 16) {
            int m = lane;
            int e = ebase + mt * 16 + m;
            float y  = (e < E_POS) ? 0.f : ((e < 2 * E_POS) ? 1.f : 2.f);
            int  pe  = perm[e];
            float yp = (pe < E_POS) ? 0.f : ((pe < 2 * E_POS) ? 1.f : 2.f);
            float dy = yp - y;
            float pr = 0.f, sh = 0.f;
            for (int n = 0; n < 130; ++n) {
                float base = Os[wave][m][n];
                float bb   = b1[n];
                float w1y  = info_w1[n * 131 + 130];
                float w    = w2[n];
                pr += w * fmaxf(base + bb, 0.f);
                sh += w * fmaxf(base + dy * w1y + bb, 0.f);
            }
            pr += b2[0]; sh += b2[0];
            sp = pr;
            se = expf(sh);
        }
        #pragma unroll
        for (int off = 16; off > 0; off >>= 1) {
            sp += __shfl_down(sp, off, 32);
            se += __shfl_down(se, off, 32);
        }
        if (lane == 0) {
            partial_pred[wtile * MT + mt] = sp;
            partial_exp [wtile * MT + mt] = se;
        }
    }
}

// ---------------- 6) final deterministic combine ----------------
__global__ __launch_bounds__(256) void finalize_kernel(
        const float* p0, const float* p1, const float* p2, const float* p3,
        const float* p4, const float* p5, const float* p6,
        int n0, int n1, int n2, int n3, int n4, int n5, int n6,
        float* __restrict__ out) {
    __shared__ float red[256];
    __shared__ float sums[7];
    const float* ps[7] = {p0, p1, p2, p3, p4, p5, p6};
    int ns[7] = {n0, n1, n2, n3, n4, n5, n6};
    for (int a = 0; a < 7; ++a) {
        float s = 0.f;
        for (int t = threadIdx.x; t < ns[a]; t += 256) s += ps[a][t];
        red[threadIdx.x] = s;
        __syncthreads();
        for (int st = 128; st > 0; st >>= 1) {
            if (threadIdx.x < st) red[threadIdx.x] += red[threadIdx.x + st];
            __syncthreads();
        }
        if (threadIdx.x == 0) sums[a] = red[0];
        __syncthreads();
    }
    if (threadIdx.x == 0) {
        float nll = (sums[0] / (float)E_POS + sums[1] / (float)E_NEG +
                     sums[2] / (float)E_NONE) / 3.f;
        float l1  = sums[3] / (float)E_POS;
        float l2  = sums[4] / (float)E_NEG;
        float l3  = -(sums[5] / (float)E_TOT - logf(sums[6] / (float)E_TOT));
        out[0] = nll + LAMBDA_S * (l1 + l2) + GAMMA * l3;
    }
}

// ---------------- launch ----------------
extern "C" void kernel_launch(void* const* d_in, const int* in_sizes, int n_in,
                              void* d_out, int out_size, void* d_ws, size_t ws_size,
                              hipStream_t stream) {
    const float* z       = (const float*)d_in[0];
    const float* lin_w   = (const float*)d_in[1];
    const float* lin_b   = (const float*)d_in[2];
    const float* info_w1 = (const float*)d_in[3];
    const float* info_b1 = (const float*)d_in[4];
    const float* info_w2 = (const float*)d_in[5];
    const float* info_b2 = (const float*)d_in[6];
    const int* pos   = (const int*)d_in[7];
    const int* neg   = (const int*)d_in[8];
    const int* none  = (const int*)d_in[9];
    const int* none2 = (const int*)d_in[10];
    const int* pos_k = (const int*)d_in[11];
    const int* neg_k = (const int*)d_in[12];
    const int* perm  = (const int*)d_in[13];

    const int NB_POS  = (E_POS  + 255) / 256;   // 977
    const int NB_NEG  = (E_NEG  + 255) / 256;   // 977
    const int NB_NONE = (E_NONE + 255) / 256;   // 1954

    char* ws = (char*)d_ws;
    size_t cur = 0;
    auto take = [&](size_t bytes) -> char* {
        cur = (cur + 255) & ~(size_t)255;
        char* p = ws + cur;
        cur += bytes;
        return p;
    };
    float*          xf     = (float*)         take((size_t)N_NODES * XD * 4);
    unsigned short* xbf    = (unsigned short*)take((size_t)N_NODES * XD * 2);
    unsigned short* w1p    = (unsigned short*)take((size_t)NPAD * KPAD * 2);
    float* p_nll0 = (float*)take((size_t)NB_POS  * 4);
    float* p_nll1 = (float*)take((size_t)NB_NEG  * 4);
    float* p_nll2 = (float*)take((size_t)NB_NONE * 4);
    float* p_l1   = (float*)take((size_t)NB_POS  * 4);
    float* p_l2   = (float*)take((size_t)NB_NEG  * 4);
    float* p_pred = (float*)take((size_t)NSUB    * 4);
    float* p_exp  = (float*)take((size_t)NSUB    * 4);
    (void)ws_size; (void)in_sizes; (void)n_in; (void)out_size;

    expmap_kernel<<<(N_NODES + 7) / 8, 256, 0, stream>>>(z, xf, xbf);
    w1prep_kernel<<<(NPAD * KPAD + 255) / 256, 256, 0, stream>>>(info_w1, w1p);

    nll_kernel<<<NB_POS,  256, 0, stream>>>(z, lin_w, lin_b, pos,  E_POS,  0, p_nll0);
    nll_kernel<<<NB_NEG,  256, 0, stream>>>(z, lin_w, lin_b, neg,  E_NEG,  1, p_nll1);
    nll_kernel<<<NB_NONE, 256, 0, stream>>>(z, lin_w, lin_b, none, E_NONE, 2, p_nll2);

    triplet_kernel<<<NB_POS, 256, 0, stream>>>(xf, pos, pos_k, E_POS,  1.f, p_l1);
    triplet_kernel<<<NB_NEG, 256, 0, stream>>>(xf, neg, neg_k, E_NEG, -1.f, p_l2);

    info_kernel<<<INFO_BLOCKS, 32 * WAVES, 0, stream>>>(
        xbf, w1p, info_w1, info_b1, info_w2, info_b2,
        pos, neg, none2, perm, p_pred, p_exp);

    finalize_kernel<<<1, 256, 0, stream>>>(
        p_nll0, p_nll1, p_nll2, p_l1, p_l2, p_pred, p_exp,
        NB_POS, NB_NEG, NB_NONE, NB_POS, NB_NEG, NSUB, NSUB,
        (float*)d_out);
}
